// ACA_80771154968927
// MI455X (gfx1250) — matrix operand order repaired and checked by reading, not compiled, not run on hardware
//
#include <hip/hip_runtime.h>

typedef __attribute__((ext_vector_type(16))) __bf16 v16bf;
typedef __attribute__((ext_vector_type(8)))  float  v8f;
typedef unsigned int v4u __attribute__((ext_vector_type(4)));
typedef int          v8i __attribute__((ext_vector_type(8)));
typedef int          v4i __attribute__((ext_vector_type(4)));

#define B_   32
#define C_   256
#define H_   80
#define W_   80
#define CH_  128
#define HW_  (H_*W_)
#define KD_  384          /* 3*CH_ */
#define EPS_ 1e-3f

#if defined(__has_builtin)
#if __has_builtin(__builtin_amdgcn_tensor_load_to_lds) && \
    __has_builtin(__builtin_amdgcn_s_wait_tensorcnt)
#define USE_TDM 1
#endif
#endif

__device__ __forceinline__ float sigf(float t)  { return 1.0f/(1.0f+__expf(-t)); }
__device__ __forceinline__ float siluf(float t) { return t*sigf(t); }

#ifdef USE_TDM
/* workgroup-relative LDS byte offset of a __shared__ pointer */
__device__ __forceinline__ unsigned lds_addr_of(const void* p) {
  return (unsigned)(unsigned long long)(const __attribute__((address_space(3))) void*)p;
}

/* Issue a TDM 2-D tile load (bf16 elements) global -> LDS.
   D# packing per CDNA5 ISA:
   group0: [1:0]=count=1, [63:32]=lds_addr, [120:64]=global_addr, [127:126]=type=2
   group1: [17:16]=data_size(1 -> 2B), [79:48]=tensor_dim0, [111:80]=tensor_dim1,
           [127:112]=tile_dim0, [143:128]=tile_dim1, [207:160]=tensor_dim0_stride */
__device__ __forceinline__ void tdm_load_2d_bf16(const __bf16* gsrc, unsigned lds_off,
                                                 unsigned tile0, unsigned tile1,
                                                 unsigned tdim0, unsigned tdim1,
                                                 unsigned stride0_elems)
{
  unsigned long long ga = (unsigned long long)gsrc;
  v4u g0;
  g0[0] = 1u;                                            /* count=1, user D#   */
  g0[1] = lds_off;                                       /* LDS byte address   */
  g0[2] = (unsigned)(ga & 0xFFFFFFFFu);                  /* global_addr[31:0]  */
  g0[3] = (unsigned)((ga >> 32) & 0x01FFFFFFu) | (2u<<30);/* [56:32] + type=2  */
  v8i g1;
  g1[0] = (int)(1u << 16);                               /* data_size=1 (2B)   */
  g1[1] = (int)((tdim0 & 0xFFFFu) << 16);                /* tensor_dim0[15:0]  */
  g1[2] = (int)(((tdim0 >> 16) & 0xFFFFu) |
                ((tdim1 & 0xFFFFu) << 16));              /* dim0 hi | dim1 lo  */
  g1[3] = (int)(((tdim1 >> 16) & 0xFFFFu) |
                ((tile0 & 0xFFFFu) << 16));              /* dim1 hi | tile0    */
  g1[4] = (int)(tile1 & 0xFFFFu);                        /* tile1, tile2=0     */
  g1[5] = (int)stride0_elems;                            /* dim0_stride[31:0]  */
  g1[6] = 0;                                             /* stride hi, dim1str */
  g1[7] = 0;
  v4i z4 = {0,0,0,0};
  v8i z8 = {0,0,0,0,0,0,0,0};
  /* this toolchain declares the 6-arg form:
     (v4u g0, v8i g1, v4i g2, v4i g3, v8i g4, int cpol) */
  __builtin_amdgcn_tensor_load_to_lds(g0, g1, z4, z4, z8, 0);
}
#endif

/* ---------------- kernel 0: convert wo (256x384 f32) to bf16 ---------------- */
__global__ void aca_wcvt(const float* __restrict__ wo, __bf16* __restrict__ wob) {
  int i = blockIdx.x*blockDim.x + threadIdx.x;
  if (i < C_*KD_) wob[i] = (__bf16)wo[i];
}

/* ------------- kernel 1: grouped convs + BN + SiLU -> bf16 q,k,v ------------ */
__global__ void aca_prep(const float* __restrict__ x,
                         const float* __restrict__ wq, const float* __restrict__ gq,
                         const float* __restrict__ bq, const float* __restrict__ mq,
                         const float* __restrict__ vq,
                         const float* __restrict__ wk, const float* __restrict__ gk,
                         const float* __restrict__ bk, const float* __restrict__ mk,
                         const float* __restrict__ vk,
                         const float* __restrict__ wv, const float* __restrict__ gv,
                         const float* __restrict__ bv, const float* __restrict__ mv,
                         const float* __restrict__ vv,
                         __bf16* __restrict__ qb, __bf16* __restrict__ kb,
                         __bf16* __restrict__ vb)
{
  long idx = (long)blockIdx.x*blockDim.x + threadIdx.x;
  if (idx >= (long)B_*CH_*HW_) return;
  int w = (int)(idx % W_);
  int h = (int)((idx / W_) % H_);
  int c = (int)((idx / HW_) % CH_);
  int b = (int)(idx / ((long)HW_*CH_));

  float aq = 0.f, ak = 0.f, av = 0.f;
  #pragma unroll
  for (int j = 0; j < 2; j++) {
    const float* xp  = x + (long)(b*C_ + 2*c + j)*HW_;
    const float* wqp = wq + (c*2+j)*3;
    const float* wkp = wk + (c*2+j)*3;
    const float* wvp = wv + (c*2+j)*9;
    #pragma unroll
    for (int t = 0; t < 3; t++) {
      int ww = w - 2 + t;                       /* q: left-pad 2 along W */
      if (ww >= 0) aq += xp[h*W_ + ww] * wqp[t];
      int hh = h - 2 + t;                       /* k: top-pad 2 along H */
      if (hh >= 0) ak += xp[hh*W_ + w] * wkp[t];
    }
    #pragma unroll
    for (int dy = 0; dy < 3; dy++) {            /* v: bottom/right pad 2 */
      int hh = h + dy; if (hh >= H_) continue;
      #pragma unroll
      for (int dx = 0; dx < 3; dx++) {
        int ww = w + dx; if (ww >= W_) continue;
        av += xp[hh*W_ + ww] * wvp[dy*3+dx];
      }
    }
  }
  float yq = (aq - mq[c]) * (gq[c]*rsqrtf(vq[c]+EPS_)) + bq[c];
  float yk = (ak - mk[c]) * (gk[c]*rsqrtf(vk[c]+EPS_)) + bk[c];
  float yv = (av - mv[c]) * (gv[c]*rsqrtf(vv[c]+EPS_)) + bv[c];
  qb[idx] = (__bf16)siluf(yq);
  kb[idx] = (__bf16)siluf(yk);
  vb[idx] = (__bf16)siluf(yv);
}

/* --------- kernel 2: per-(b,c) attention: QK^T -> flat softmax -> PV -------- */
__global__ void __launch_bounds__(256) aca_attn(const __bf16* __restrict__ qb,
                                                const __bf16* __restrict__ kb,
                                                const __bf16* __restrict__ vb,
                                                __bf16* __restrict__ ob)
{
  __shared__ __bf16 sQ[HW_];
  __shared__ __bf16 sK[HW_];
  __shared__ __bf16 sV[HW_];
  __shared__ float  sS[HW_];
  __shared__ float  red[256];

  int tid  = threadIdx.x;
  int wave = tid >> 5, lane = tid & 31;
  int ln16 = lane & 15, half = lane >> 4;
  long base = (long)blockIdx.x * HW_;          /* blockIdx.x == b*CH_ + c */

#ifdef USE_TDM
  if (tid < 32) {                              /* wave 0 drives the TDM */
    tdm_load_2d_bf16(qb+base, lds_addr_of(sQ), HW_, 1, HW_, 1, HW_);
    tdm_load_2d_bf16(kb+base, lds_addr_of(sK), HW_, 1, HW_, 1, HW_);
    tdm_load_2d_bf16(vb+base, lds_addr_of(sV), HW_, 1, HW_, 1, HW_);
    __builtin_amdgcn_s_wait_tensorcnt(0);
  }
#else
  for (int i = tid; i < HW_; i += 256) {
    sQ[i] = qb[base+i]; sK[i] = kb[base+i]; sV[i] = vb[base+i];
  }
#endif
  __syncthreads();

  /* ---- S = Q K^T / sqrt(128), 25 tiles of 16x16, K=80 padded to 96 ---- */
  const float scale = 0.08838834764831845f;    /* 1/sqrt(128) */
  for (int tile = wave; tile < 25; tile += 8) {
    int ti = tile/5, tj = tile%5;
    v8f acc = {};
    #pragma unroll
    for (int ks = 0; ks < 80; ks += 32) {
      v16bf a, bfr;
      #pragma unroll
      for (int e = 0; e < 16; e++) {
        int ka = ks + e + 16*half;                         /* A lane layout */
        a[e]   = (ka < 80) ? sQ[(ti*16 + ln16)*80 + ka] : (__bf16)0.0f;
        int kb_ = ks + e + 16*half;                        /* B lane layout */
        bfr[e] = (kb_ < 80) ? sK[(tj*16 + ln16)*80 + kb_] : (__bf16)0.0f;
      }
      acc = __builtin_amdgcn_wmma_f32_16x16x32_bf16(false, a, false, bfr,
                                                    (short)0, acc, false, false);
    }
    #pragma unroll
    for (int r = 0; r < 8; r++)
      sS[(ti*16 + r + 8*half)*80 + tj*16 + ln16] = acc[r]*scale;
  }
  __syncthreads();

  /* ---- softmax over the flattened 6400-element block ---- */
  float mx = -1e30f;
  for (int i = tid; i < HW_; i += 256) mx = fmaxf(mx, sS[i]);
  red[tid] = mx; __syncthreads();
  for (int s = 128; s > 0; s >>= 1) {
    if (tid < s) red[tid] = fmaxf(red[tid], red[tid+s]);
    __syncthreads();
  }
  float gmax = red[0]; __syncthreads();
  float sm = 0.f;
  for (int i = tid; i < HW_; i += 256) {
    float p = __expf(sS[i] - gmax); sS[i] = p; sm += p;
  }
  red[tid] = sm; __syncthreads();
  for (int s = 128; s > 0; s >>= 1) {
    if (tid < s) red[tid] += red[tid+s];
    __syncthreads();
  }
  float invZ = 1.0f/red[0];

  /* ---- O = P @ V (contraction over j), scaled by 1/Z ---- */
  for (int tile = wave; tile < 25; tile += 8) {
    int ti = tile/5, tw = tile%5;
    v8f acc = {};
    #pragma unroll
    for (int ks = 0; ks < 80; ks += 32) {
      v16bf a, bfr;
      #pragma unroll
      for (int e = 0; e < 16; e++) {
        int ka = ks + e + 16*half;
        a[e]   = (ka < 80) ? (__bf16)sS[(ti*16 + ln16)*80 + ka] : (__bf16)0.0f;
        int kb_ = ks + e + 16*half;
        bfr[e] = (kb_ < 80) ? sV[kb_*80 + tw*16 + ln16] : (__bf16)0.0f;
      }
      acc = __builtin_amdgcn_wmma_f32_16x16x32_bf16(false, a, false, bfr,
                                                    (short)0, acc, false, false);
    }
    #pragma unroll
    for (int r = 0; r < 8; r++)
      ob[base + (ti*16 + r + 8*half)*80 + tw*16 + ln16] = (__bf16)(acc[r]*invZ);
  }
}

/* -------- kernel 3: 1x1 conv GEMM [co=256, k=384, px] + BN/SiLU + gate ------ */
#define PIX_TILE 64
#define CO_TILE  128
#define NKS      (KD_/32)   /* 12 K-steps */

__global__ void __launch_bounds__(256) aca_out(
    const __bf16* __restrict__ ob, const __bf16* __restrict__ qb,
    const __bf16* __restrict__ kb, const __bf16* __restrict__ wob,
    const float* __restrict__ go, const float* __restrict__ bo,
    const float* __restrict__ mo, const float* __restrict__ vo,
    const float* __restrict__ x, float* __restrict__ out)
{
  __shared__ __bf16 sB[2][32*PIX_TILE];        /* double-buffered B tile      */
  int tid  = threadIdx.x;
  int wave = tid >> 5, lane = tid & 31;
  int ln16 = lane & 15, half = lane >> 4;
  int wm = wave & 3;                            /* 4 waves along co  (4*32=128) */
  int wn = wave >> 2;                           /* 2 waves along px  (2*32=64)  */
  int b   = blockIdx.z;
  int co0 = blockIdx.y * CO_TILE;
  int p0  = blockIdx.x * PIX_TILE;

#ifdef USE_TDM
  /* cat = [attn_out, q, k]; each 32-channel K-step lies inside one plane */
  auto tdm_issue = [&](int s, int buf) {
    int ci0 = s*32;
    const __bf16* plane = (ci0 < CH_) ? ob : (ci0 < 2*CH_) ? qb : kb;
    int cc0 = ci0 & (CH_-1);
    const __bf16* g = plane + (long)(b*CH_ + cc0)*HW_ + p0;
    /* 2-D tile: 64 px (dim0) x 32 channels (dim1), row stride HW_ elems */
    tdm_load_2d_bf16(g, lds_addr_of(&sB[buf][0]), PIX_TILE, 32,
                     PIX_TILE, 32, HW_);
  };
  if (tid < 32) tdm_issue(0, 0);
#endif

  v8f acc[2][2] = {};
  for (int s = 0; s < NKS; s++) {
    int ks = s*32;
#ifdef USE_TDM
    if (tid < 32) {
      if (s+1 < NKS) { tdm_issue(s+1, (s+1)&1);
                       __builtin_amdgcn_s_wait_tensorcnt(1); }
      else           { __builtin_amdgcn_s_wait_tensorcnt(0); }
    }
    __syncthreads();
    const __bf16* bt = &sB[s&1][0];
#else
    __syncthreads();
    for (int i = tid; i < 32*PIX_TILE; i += 256) {
      int kr = i >> 6, px = i & 63;
      int ci = ks + kr;
      const __bf16* plane = (ci < CH_) ? ob : (ci < 2*CH_) ? qb : kb;
      int cc = ci & (CH_-1);
      sB[0][i] = plane[(long)(b*CH_ + cc)*HW_ + p0 + px];
    }
    __syncthreads();
    const __bf16* bt = &sB[0][0];
#endif
    #pragma unroll
    for (int mt = 0; mt < 2; mt++) {
      v16bf a;
      int m = co0 + wm*32 + mt*16 + ln16;
      #pragma unroll
      for (int e = 0; e < 16; e++) {
        int k = e + 16*half;
        a[e] = wob[m*KD_ + ks + k];
      }
      #pragma unroll
      for (int nt = 0; nt < 2; nt++) {
        v16bf bf;
        #pragma unroll
        for (int e = 0; e < 16; e++) {
          int k = e + 16*half;
          bf[e] = bt[k*PIX_TILE + wn*32 + nt*16 + ln16];
        }
        acc[mt][nt] = __builtin_amdgcn_wmma_f32_16x16x32_bf16(
            false, a, false, bf, (short)0, acc[mt][nt], false, false);
      }
    }
#ifdef USE_TDM
    __syncthreads();   /* all waves done with sB[s&1] before it is re-filled */
#endif
  }
  /* epilogue: BN + SiLU + x*sigmoid(out) */
  #pragma unroll
  for (int mt = 0; mt < 2; mt++) {
    #pragma unroll
    for (int nt = 0; nt < 2; nt++) {
      #pragma unroll
      for (int r = 0; r < 8; r++) {
        int co = co0 + wm*32 + mt*16 + r + 8*half;
        int px = p0  + wn*32 + nt*16 + ln16;
        float y = (acc[mt][nt][r] - mo[co]) * (go[co]*rsqrtf(vo[co]+EPS_)) + bo[co];
        float s = siluf(y);
        long xi = (long)(b*C_ + co)*HW_ + px;
        out[xi] = x[xi] * sigf(s);
      }
    }
  }
}

/* --------------------------------- launch ---------------------------------- */
extern "C" void kernel_launch(void* const* d_in, const int* in_sizes, int n_in,
                              void* d_out, int out_size, void* d_ws, size_t ws_size,
                              hipStream_t stream)
{
  (void)in_sizes; (void)n_in; (void)out_size; (void)ws_size;
  const float* x  = (const float*)d_in[0];
  const float* wq = (const float*)d_in[1];
  const float* gq = (const float*)d_in[2];
  const float* bq = (const float*)d_in[3];
  const float* mq = (const float*)d_in[4];
  const float* vq = (const float*)d_in[5];
  const float* wk = (const float*)d_in[6];
  const float* gk = (const float*)d_in[7];
  const float* bk = (const float*)d_in[8];
  const float* mk = (const float*)d_in[9];
  const float* vk = (const float*)d_in[10];
  const float* wv = (const float*)d_in[11];
  const float* gv = (const float*)d_in[12];
  const float* bv = (const float*)d_in[13];
  const float* mv = (const float*)d_in[14];
  const float* vv = (const float*)d_in[15];
  const float* wo = (const float*)d_in[16];
  const float* go = (const float*)d_in[17];
  const float* bo = (const float*)d_in[18];
  const float* mo = (const float*)d_in[19];
  const float* vo = (const float*)d_in[20];

  char* ws = (char*)d_ws;
  const size_t planeBytes = (size_t)B_*CH_*HW_*sizeof(__bf16); /* 52,428,800 */
  __bf16* qb  = (__bf16*)(ws);
  __bf16* kb  = (__bf16*)(ws + 1*planeBytes);
  __bf16* vb  = (__bf16*)(ws + 2*planeBytes);
  __bf16* obp = (__bf16*)(ws + 3*planeBytes);
  __bf16* wob = (__bf16*)(ws + 4*planeBytes);

  aca_wcvt<<<(C_*KD_ + 255)/256, 256, 0, stream>>>(wo, wob);

  long n = (long)B_*CH_*HW_;
  aca_prep<<<(unsigned)((n + 255)/256), 256, 0, stream>>>(
      x, wq,gq,bq,mq,vq, wk,gk,bk,mk,vk, wv,gv,bv,mv,vv, qb,kb,vb);

  aca_attn<<<B_*CH_, 256, 0, stream>>>(qb, kb, vb, obp);

  dim3 g(HW_/PIX_TILE, C_/CO_TILE, B_);
  aca_out<<<g, 256, 0, stream>>>(obp, qb, kb, wob, go, bo, mo, vo, x,
                                 (float*)d_out);
}
